// SGL_8461085573266
// MI455X (gfx1250) — matrix operand order repaired and checked by reading, not compiled, not run on hardware
//
#include <hip/hip_runtime.h>
#include <math.h>

#define NUM_USERS 100000
#define NUM_ITEMS 50000
#define N_NODES   150000
#define EMB       64
#define BATCH     4096
#define INV_T     5.0f                      // 1 / 0.2
#define INV_T_LOG2E 7.2134752044448170f     // (1/0.2) * log2(e): fold temp + exp2 conversion

typedef __attribute__((ext_vector_type(16))) _Float16 v16h;
typedef __attribute__((ext_vector_type(8)))  float    v8f;

static const int ND = N_NODES * EMB;   // 9,600,000 floats per big buffer
static const int BD = BATCH * EMB;     // 262,144

// ---------------- graph propagation kernels ----------------

__global__ void k_init(const float* __restrict__ ue, const float* __restrict__ ie,
                       float* __restrict__ acc, float* __restrict__ cur, int n) {
    int i = blockIdx.x * blockDim.x + threadIdx.x;
    if (i >= n) return;
    float v = (i < NUM_USERS * EMB) ? ue[i] : ie[i - NUM_USERS * EMB];
    acc[i] = v;
    cur[i] = v;
}

__global__ void k_zero(float* __restrict__ p, int n) {
    int i = blockIdx.x * blockDim.x + threadIdx.x;
    if (i < n) p[i] = 0.0f;
}

__global__ void k_axpy(float* __restrict__ acc, const float* __restrict__ nxt, int n) {
    int i = blockIdx.x * blockDim.x + threadIdx.x;
    if (i < n) acc[i] += nxt[i];
}

// edge-parallel SpMM: 16 threads per edge, each covers 4 contiguous dims.
// Random 256B row gathers + f32 scatter-atomics; 38MB tables stay L2-resident.
__global__ void k_spmm(const int* __restrict__ rows, const int* __restrict__ cols,
                       const float* __restrict__ vals, const float* __restrict__ x,
                       float* __restrict__ y, int E) {
    int t = blockIdx.x * blockDim.x + threadIdx.x;
    if (t >= E * 16) return;
    int e = t >> 4;
    int d = (t & 15) << 2;
    int row = rows[e];
    int col = cols[e];
    float v  = vals[e];
    const float4 xv = *(const float4*)(x + col * 64 + d);
    float* yp = y + row * 64 + d;
    atomicAdd(yp + 0, v * xv.x);
    atomicAdd(yp + 1, v * xv.y);
    atomicAdd(yp + 2, v * xv.z);
    atomicAdd(yp + 3, v * xv.w);
}

// ---------------- gathers ----------------

// final = acc / 4 applied here (layer-mean)
__global__ void k_gather3(const float* __restrict__ acc,
                          const int* __restrict__ user, const int* __restrict__ pos,
                          const int* __restrict__ neg,
                          float* __restrict__ ue, float* __restrict__ pe,
                          float* __restrict__ ne) {
    int t = blockIdx.x * blockDim.x + threadIdx.x;
    if (t >= BATCH * EMB) return;
    int i = t >> 6, d = t & 63;
    ue[t] = acc[user[i] * 64 + d] * 0.25f;
    pe[t] = acc[(NUM_USERS + pos[i]) * 64 + d] * 0.25f;
    ne[t] = acc[(NUM_USERS + neg[i]) * 64 + d] * 0.25f;
}

// gather + L2-normalize + convert to f16 (normalization absorbs the /4 scale)
__global__ void k_gather_norm(const float* __restrict__ acc, const int* __restrict__ idx,
                              int base, _Float16* __restrict__ out) {
    int i = blockIdx.x * blockDim.x + threadIdx.x;
    if (i >= BATCH) return;
    const float* rp = acc + (long)(base + idx[i]) * 64;
    float ss = 0.0f;
    #pragma unroll
    for (int d = 0; d < 64; ++d) { float v = rp[d]; ss += v * v; }
    float inv = rsqrtf(ss + 1e-20f);
    _Float16* op = out + (long)i * 64;
    #pragma unroll
    for (int d = 0; d < 64; ++d) op[d] = (_Float16)(rp[d] * inv);
}

// positive-pair similarity (diagonal), already divided by temperature
__global__ void k_pos(const _Float16* __restrict__ a, const _Float16* __restrict__ b,
                      float* __restrict__ pos) {
    int i = blockIdx.x * blockDim.x + threadIdx.x;
    if (i >= BATCH) return;
    const _Float16* ap = a + (long)i * 64;
    const _Float16* bp = b + (long)i * 64;
    float s = 0.0f;
    #pragma unroll
    for (int d = 0; d < 64; ++d) s += (float)ap[d] * (float)bp[d];
    pos[i] = s * INV_T;
}

// ---------------- fused WMMA GEMM + logsumexp (InfoNCE) ----------------
// One wave per 16-row strip of the 4096x4096 logits. A held in registers,
// B streamed tile-by-tile; per-tile 16x16 f32 accum -> exp2 -> per-row sums.
__global__ void __launch_bounds__(256) k_infonce(const _Float16* __restrict__ v1,
                                                 const _Float16* __restrict__ v2,
                                                 const float* __restrict__ pos,
                                                 float* __restrict__ accum) {
    int wave = (blockIdx.x * blockDim.x + threadIdx.x) >> 5;
    int lane = threadIdx.x & 31;
    int m0   = wave * 16;
    int r    = lane & 15;
    int hi   = lane >> 4;

    // A-matrix 16x32 f16 layout: lane<16 -> K=0..15 ; lane>=16 -> K=16..31 (contiguous, matches B)
    const _Float16* ap = v1 + (m0 + r) * 64;
    v16h a0, a1;
    #pragma unroll
    for (int e = 0; e < 16; ++e) {
        int ko = hi * 16 + e;
        a0[e] = ap[ko];
        a1[e] = ap[32 + ko];
    }

    float partial[8];
    #pragma unroll
    for (int v = 0; v < 8; ++v) partial[v] = 0.0f;

    for (int n0 = 0; n0 < BATCH; n0 += 16) {
        // B-matrix 32x16 f16: lane holds column n0+r, K = kc + hi*16 + e (contiguous, 32B-aligned)
        const _Float16* bp = v2 + (n0 + r) * 64 + hi * 16;
        v16h b0 = *(const v16h*)(bp);        // K chunk 0..31
        v16h b1 = *(const v16h*)(bp + 32);   // K chunk 32..63
        v8f c = {};
        c = __builtin_amdgcn_wmma_f32_16x16x32_f16(false, a0, false, b0, (short)0, c, false, false);
        c = __builtin_amdgcn_wmma_f32_16x16x32_f16(false, a1, false, b1, (short)0, c, false, false);
        // C/D layout: lane column = r, VGPR v = row (v + 8*hi).
        // exp(l/T) == exp2(l * INV_T_LOG2E); |arg| <= ~7.3 so the raw v_exp_f32
        // (no subnormal range fixup) is exact: use the bare hardware intrinsic.
        #pragma unroll
        for (int v = 0; v < 8; ++v)
            partial[v] += __builtin_amdgcn_exp2f(c[v] * INV_T_LOG2E);
    }

    // reduce the 16 columns held across lanes of each half-wave
    #pragma unroll
    for (int v = 0; v < 8; ++v) {
        float s = partial[v];
        s += __shfl_xor(s, 1, 32);
        s += __shfl_xor(s, 2, 32);
        s += __shfl_xor(s, 4, 32);
        s += __shfl_xor(s, 8, 32);
        partial[v] = s;
    }

    if (r == 0) {   // lane 0 -> rows m0..m0+7, lane 16 -> rows m0+8..m0+15
        float local = 0.0f;
        #pragma unroll
        for (int v = 0; v < 8; ++v) {
            int row = m0 + hi * 8 + v;
            local += __logf(partial[v]) - pos[row];
        }
        atomicAdd(accum, local);
    }
}

// ---------------- BPR + regularization ----------------

__global__ void k_bpr_reg(const float* __restrict__ ue, const float* __restrict__ pe,
                          const float* __restrict__ ne,
                          const float* __restrict__ uemb, const float* __restrict__ iemb,
                          const int* __restrict__ user, const int* __restrict__ posi,
                          const int* __restrict__ negi, float* __restrict__ accum) {
    int i = blockIdx.x * blockDim.x + threadIdx.x;
    if (i >= BATCH) return;
    const float* u = ue + (long)i * 64;
    const float* p = pe + (long)i * 64;
    const float* n = ne + (long)i * 64;
    float ps = 0.0f, ns = 0.0f;
    #pragma unroll
    for (int d = 0; d < 64; ++d) { ps += u[d] * p[d]; ns += u[d] * n[d]; }
    float x  = ns - ps;
    float sp = fmaxf(x, 0.0f) + log1pf(__expf(-fabsf(x)));   // softplus, stable

    const float* eu = uemb + (long)user[i] * 64;
    const float* ep = iemb + (long)posi[i] * 64;
    const float* en = iemb + (long)negi[i] * 64;
    float rg = 0.0f;
    #pragma unroll
    for (int d = 0; d < 64; ++d)
        rg += eu[d] * eu[d] + ep[d] * ep[d] + en[d] * en[d];

    atomicAdd(accum + 0, sp);
    atomicAdd(accum + 1, rg);
}

__global__ void k_zero_accum(float* __restrict__ a) {
    if (threadIdx.x < 4) a[threadIdx.x] = 0.0f;
}

__global__ void k_finalize(const float* __restrict__ accum, float* __restrict__ out) {
    if (threadIdx.x == 0 && blockIdx.x == 0) {
        const float invB = 1.0f / (float)BATCH;
        out[0] = accum[0] * invB;                       // bpr
        out[1] = 1e-4f * 0.5f * accum[1] * invB;        // reg
        out[2] = 0.1f * (accum[2] + accum[3]) * invB;   // ssl
    }
}

// ---------------- launcher ----------------

extern "C" void kernel_launch(void* const* d_in, const int* in_sizes, int n_in,
                              void* d_out, int out_size, void* d_ws, size_t ws_size,
                              hipStream_t stream) {
    const int*   user     = (const int*)d_in[0];
    const int*   positive = (const int*)d_in[1];
    const int*   negative = (const int*)d_in[2];
    const int*   g_rows[3] = {(const int*)d_in[3], (const int*)d_in[6], (const int*)d_in[9]};
    const int*   g_cols[3] = {(const int*)d_in[4], (const int*)d_in[7], (const int*)d_in[10]};
    const float* g_vals[3] = {(const float*)d_in[5], (const float*)d_in[8], (const float*)d_in[11]};
    const int    g_E[3]    = {in_sizes[3], in_sizes[6], in_sizes[9]};
    const float* uemb = (const float*)d_in[12];
    const float* iemb = (const float*)d_in[13];

    float* ws  = (float*)d_ws;
    float* acc = ws;
    float* cur = ws + (size_t)ND;
    float* nxt = ws + 2 * (size_t)ND;
    float* ue  = ws + 3 * (size_t)ND;
    float* pe  = ue + BD;
    float* ne  = pe + BD;
    _Float16* hbase = (_Float16*)(ne + BD);
    _Float16* u1n = hbase;
    _Float16* u2n = hbase + (size_t)BD;
    _Float16* i1n = hbase + 2 * (size_t)BD;
    _Float16* i2n = hbase + 3 * (size_t)BD;
    float* pos_u = (float*)(hbase + 4 * (size_t)BD);
    float* pos_i = pos_u + BATCH;
    float* accum = pos_i + BATCH;

    const int TB = 256;
    const int ndBlocks = (ND + TB - 1) / TB;

    k_zero_accum<<<1, 32, 0, stream>>>(accum);

    for (int g = 0; g < 3; ++g) {
        k_init<<<ndBlocks, TB, 0, stream>>>(uemb, iemb, acc, cur, ND);
        float* c  = cur;
        float* nx = nxt;
        for (int l = 0; l < 3; ++l) {
            k_zero<<<ndBlocks, TB, 0, stream>>>(nx, ND);
            int tot = g_E[g] * 16;
            k_spmm<<<(tot + TB - 1) / TB, TB, 0, stream>>>(g_rows[g], g_cols[g], g_vals[g],
                                                           c, nx, g_E[g]);
            k_axpy<<<ndBlocks, TB, 0, stream>>>(acc, nx, ND);
            float* t = c; c = nx; nx = t;
        }
        if (g == 0) {
            k_gather3<<<(BATCH * EMB + TB - 1) / TB, TB, 0, stream>>>(
                acc, user, positive, negative, ue, pe, ne);
        } else if (g == 1) {
            k_gather_norm<<<(BATCH + 63) / 64, 64, 0, stream>>>(acc, user, 0, u1n);
            k_gather_norm<<<(BATCH + 63) / 64, 64, 0, stream>>>(acc, positive, NUM_USERS, i1n);
        } else {
            k_gather_norm<<<(BATCH + 63) / 64, 64, 0, stream>>>(acc, user, 0, u2n);
            k_gather_norm<<<(BATCH + 63) / 64, 64, 0, stream>>>(acc, positive, NUM_USERS, i2n);
        }
    }

    k_pos<<<(BATCH + 63) / 64, 64, 0, stream>>>(u1n, u2n, pos_u);
    k_pos<<<(BATCH + 63) / 64, 64, 0, stream>>>(i1n, i2n, pos_i);

    // 256 strips of 16 rows -> 256 waves -> 32 blocks * 8 waves
    k_infonce<<<32, 256, 0, stream>>>(u1n, u2n, pos_u, accum + 2);
    k_infonce<<<32, 256, 0, stream>>>(i1n, i2n, pos_i, accum + 3);

    k_bpr_reg<<<(BATCH + TB - 1) / TB, TB, 0, stream>>>(ue, pe, ne, uemb, iemb,
                                                        user, positive, negative, accum);
    k_finalize<<<1, 1, 0, stream>>>(accum, (float*)d_out);
}
